// VectorVanillaTransformer_84859963835157
// MI455X (gfx1250) — matrix operand order repaired and checked by reading, not compiled, not run on hardware
//
#include <hip/hip_runtime.h>
#include <hip/hip_bf16.h>

// ---------------- constants (match reference) ----------------
constexpr int Ec   = 512;
constexpr int Hc   = 8;
constexpr int HDc  = 64;     // Ec / Hc
constexpr int D3c  = 192;    // 3 * HDc
constexpr int FFNc = 2048;
constexpr int Lc   = 4;
constexpr int Bc   = 4;
constexpr int Sc   = 1024;

typedef __attribute__((ext_vector_type(16))) _Float16 v16h;
typedef __attribute__((ext_vector_type(8)))  _Float16 v8h;
typedef __attribute__((ext_vector_type(8)))  float    v8f;
typedef __attribute__((ext_vector_type(4)))  float    v4f;

__device__ __forceinline__ v8f wmma16(v16h a, v16h b, v8f c) {
  // D = A(16x32 f16) * B(32x16 f16) + C(16x16 f32)
  return __builtin_amdgcn_wmma_f32_16x16x32_f16(
      /*neg_a=*/false, a, /*neg_b=*/false, b,
      /*c_mod=*/(short)0, c, /*reuse_a=*/false, /*reuse_b=*/false);
}

// ---------------- generic WMMA GEMM:  C[M,N] = A[M,K] * W[N,K]^T ----------------
// grid = (M/64, N/64), block = 128 (4 waves). M%64==0, N%64==0, K%32==0.
__global__ __launch_bounds__(128)
void k_gemm(const _Float16* __restrict__ A, const _Float16* __restrict__ W,
            float* __restrict__ C, int K, int ldc) {
  __shared__ _Float16 As[64 * 40];   // pitch 40 halves = 80 B (16 B aligned)
  __shared__ _Float16 Bs[64 * 40];

  const int m0   = blockIdx.x * 64;
  const int n0   = blockIdx.y * 64;
  const int tid  = threadIdx.x;
  const int wid  = tid >> 5;
  const int lane = tid & 31;
  const int hl   = lane & 15;
  const int hi   = lane >> 4;

  v8f acc[4];
  #pragma unroll
  for (int nt = 0; nt < 4; ++nt) acc[nt] = v8f{};

  for (int k0 = 0; k0 < K; k0 += 32) {
    __syncthreads();
    // 128-bit cooperative load of 64x32 A tile and 64x32 W tile
    // 2048 halves / 8 per lane = 256 v8h units, 2 per thread
    #pragma unroll
    for (int u = tid; u < 256; u += 128) {
      const int r  = u >> 2;        // 0..63
      const int c8 = (u & 3) * 8;   // 0,8,16,24
      *(v8h*)&As[r * 40 + c8] = *(const v8h*)&A[(size_t)(m0 + r) * K + k0 + c8];
      *(v8h*)&Bs[r * 40 + c8] = *(const v8h*)&W[(size_t)(n0 + r) * K + k0 + c8];
    }
    if (k0 + 32 < K) {  // prefetch next K tile -> global_prefetch_b8
      __builtin_prefetch(&A[(size_t)(m0 + (tid >> 1)) * K + k0 + 32], 0, 0);
      __builtin_prefetch(&W[(size_t)(n0 + (tid >> 1)) * K + k0 + 32], 0, 0);
    }
    __syncthreads();

    v16h a;
    #pragma unroll
    for (int e = 0; e < 16; ++e) {
      const int kd = hi * 16 + e;   // A 16x32 layout
      a[e] = As[(wid * 16 + hl) * 40 + kd];
    }
    #pragma unroll
    for (int nt = 0; nt < 4; ++nt) {
      v16h b;
      #pragma unroll
      for (int e = 0; e < 16; ++e)                         // B 32x16 layout
        b[e] = Bs[(nt * 16 + hl) * 40 + hi * 16 + e];
      acc[nt] = wmma16(a, b, acc[nt]);
    }
  }

  #pragma unroll
  for (int nt = 0; nt < 4; ++nt)
    #pragma unroll
    for (int r = 0; r < 8; ++r)                            // C layout: M=r+8*hi, N=hl
      C[(size_t)(m0 + wid * 16 + r + hi * 8) * ldc + n0 + nt * 16 + hl] = acc[nt][r];
}

// ---------------- flash attention on vn (q==k==v), d=192 ----------------
// grid = (S/64, H, B), block = 128 (4 waves, 16 q-rows each)
__global__ __launch_bounds__(128)
void k_attn(const float* __restrict__ vn, const float* __restrict__ bias,
            const unsigned char* __restrict__ pm, _Float16* __restrict__ o16) {
  __shared__ _Float16 Qs[64 * 200];  // pitch 200 halves = 400 B (16 B aligned)
  __shared__ _Float16 Ks[64 * 200];
  __shared__ _Float16 Ps[4 * 16 * 72];

  const int q0   = blockIdx.x * 64;
  const int h    = blockIdx.y;
  const int b    = blockIdx.z;
  const int tid  = threadIdx.x;
  const int wid  = tid >> 5;
  const int lane = tid & 31;
  const int hl   = lane & 15;
  const int hi   = lane >> 4;

  // vectorized tile loader: 8 f32 -> 8 f16 per unit; 64*192/8 = 1536 units
  auto load_tile = [&](_Float16* dst, int s0) {
    for (int u = tid; u < 1536; u += 128) {
      const int i  = u / 24;         // row in tile
      const int j8 = (u % 24) * 8;   // col chunk (stays within one 64-wide slice)
      const int s  = s0 + i;
      const float* src =
          &vn[(((size_t)b * Sc + s) * 3 + (j8 >> 6)) * Ec + h * HDc + (j8 & 63)];
      const v4f f0 = *(const v4f*)src;
      const v4f f1 = *(const v4f*)(src + 4);
      v8h hv;
      #pragma unroll
      for (int e = 0; e < 4; ++e) { hv[e] = (_Float16)f0[e]; hv[4 + e] = (_Float16)f1[e]; }
      *(v8h*)&dst[i * 200 + j8] = hv;
    }
  };

  load_tile(Qs, q0);

  const float scale = 0.07216878364870322f;  // 1/sqrt(192)
  float mrow[8], lrow[8];
  v8f oacc[12];
  #pragma unroll
  for (int r = 0; r < 8; ++r) { mrow[r] = -3e30f; lrow[r] = 0.0f; }
  #pragma unroll
  for (int d = 0; d < 12; ++d) oacc[d] = v8f{};

  for (int kb = 0; kb < Sc; kb += 64) {
    __syncthreads();
    load_tile(Ks, kb);
    __syncthreads();

    // ---- scores: 16x64 tile per wave, K = 192 ----
    v8f sc[4];
    #pragma unroll
    for (int nt = 0; nt < 4; ++nt) sc[nt] = v8f{};
    for (int kk = 0; kk < 6; ++kk) {
      v16h a;
      #pragma unroll
      for (int e = 0; e < 16; ++e) {
        const int kd = kk * 32 + hi * 16 + e;
        a[e] = Qs[(wid * 16 + hl) * 200 + kd];
      }
      #pragma unroll
      for (int nt = 0; nt < 4; ++nt) {
        v16h bb;
        #pragma unroll
        for (int e = 0; e < 16; ++e)
          bb[e] = Ks[(nt * 16 + hl) * 200 + kk * 32 + hi * 16 + e];
        sc[nt] = wmma16(a, bb, sc[nt]);
      }
    }

    // ---- bias + key padding mask + online softmax ----
    float p[4][8];
    #pragma unroll
    for (int r = 0; r < 8; ++r) {
      const int q = q0 + wid * 16 + r + hi * 8;
      float mx = -3e30f;
      #pragma unroll
      for (int nt = 0; nt < 4; ++nt) {
        const int kc = kb + nt * 16 + hl;
        float sv = sc[nt][r] * scale +
                   bias[((size_t)(b * Hc + h) * Sc + q) * Sc + kc];
        if (pm[b * Sc + kc]) sv = -3e30f;
        p[nt][r] = sv;
        mx = fmaxf(mx, sv);
      }
      #pragma unroll
      for (int off = 1; off < 16; off <<= 1) mx = fmaxf(mx, __shfl_xor(mx, off, 32));
      const float mnew = fmaxf(mrow[r], mx);
      const float rs   = __expf(mrow[r] - mnew);
      mrow[r] = mnew;
      lrow[r] *= rs;
      float ls = 0.0f;
      #pragma unroll
      for (int nt = 0; nt < 4; ++nt) {
        const float pv = __expf(p[nt][r] - mnew);
        p[nt][r] = pv;
        ls += pv;
      }
      #pragma unroll
      for (int off = 1; off < 16; off <<= 1) ls += __shfl_xor(ls, off, 32);
      lrow[r] += ls;
      #pragma unroll
      for (int d = 0; d < 12; ++d) oacc[d][r] *= rs;
    }

    // ---- re-fragment P via per-wave LDS (C-layout -> A-layout) ----
    #pragma unroll
    for (int nt = 0; nt < 4; ++nt)
      #pragma unroll
      for (int r = 0; r < 8; ++r)
        Ps[wid * 1152 + (r + hi * 8) * 72 + nt * 16 + hl] = (_Float16)p[nt][r];

    // ---- O += P(16x64) * V(64x192) ----
    for (int kk = 0; kk < 2; ++kk) {
      v16h a;
      #pragma unroll
      for (int e = 0; e < 16; ++e) {
        const int kd = kk * 32 + hi * 16 + e;
        a[e] = Ps[wid * 1152 + hl * 72 + kd];
      }
      #pragma unroll
      for (int dt = 0; dt < 12; ++dt) {
        v16h bb;
        #pragma unroll
        for (int e = 0; e < 16; ++e)
          bb[e] = Ks[(kk * 32 + hi * 16 + e) * 200 + dt * 16 + hl];
        oacc[dt] = wmma16(a, bb, oacc[dt]);
      }
    }
  }

  // epilogue: o / l, scatter back to [B,S,3,E] (f16 for next GEMM)
  #pragma unroll
  for (int dt = 0; dt < 12; ++dt)
    #pragma unroll
    for (int r = 0; r < 8; ++r) {
      const int s = q0 + wid * 16 + r + hi * 8;
      const int d = dt * 16 + hl;
      const float val = oacc[dt][r] / fmaxf(lrow[r], 1e-20f);
      o16[(((size_t)b * Sc + s) * 3 + (d >> 6)) * Ec + h * HDc + (d & 63)] = (_Float16)val;
    }
}

// ---------------- layernorm over rows of length E=512 ----------------
__global__ __launch_bounds__(256)
void k_ln(const float* __restrict__ in, const float* __restrict__ w,
          const float* __restrict__ bb, float* __restrict__ out) {
  __shared__ float red[8];
  __shared__ float mv[2];
  const size_t row = blockIdx.x;
  const int tid = threadIdx.x, wid = tid >> 5, lane = tid & 31;

  const float v0 = in[row * Ec + tid];
  const float v1 = in[row * Ec + tid + 256];
  float s = v0 + v1;
  #pragma unroll
  for (int off = 16; off > 0; off >>= 1) s += __shfl_xor(s, off, 32);
  if (lane == 0) red[wid] = s;
  __syncthreads();
  if (tid == 0) { float t = 0; for (int i = 0; i < 8; ++i) t += red[i]; mv[0] = t / (float)Ec; }
  __syncthreads();
  const float mean = mv[0];
  const float d0 = v0 - mean, d1 = v1 - mean;
  float q = d0 * d0 + d1 * d1;
  #pragma unroll
  for (int off = 16; off > 0; off >>= 1) q += __shfl_xor(q, off, 32);
  if (lane == 0) red[wid] = q;
  __syncthreads();
  if (tid == 0) { float t = 0; for (int i = 0; i < 8; ++i) t += red[i]; mv[1] = rsqrtf(t / (float)Ec + 1e-5f); }
  __syncthreads();
  const float rstd = mv[1];
  out[row * Ec + tid]       = d0 * rstd * w[tid] + bb[tid];
  out[row * Ec + tid + 256] = d1 * rstd * w[tid + 256] + bb[tid + 256];
}

// ---------------- vectorized elementwise kernels ----------------
// f32 -> f16 convert, 8 elements/thread (n must be a multiple of 8)
__global__ __launch_bounds__(256)
void k_copy16(const float* __restrict__ in, _Float16* __restrict__ out, size_t n8) {
  const size_t u = (size_t)blockIdx.x * 256 + threadIdx.x;
  if (u >= n8) return;
  const float* src = in + u * 8;
  const v4f f0 = *(const v4f*)src;
  const v4f f1 = *(const v4f*)(src + 4);
  v8h o;
  #pragma unroll
  for (int e = 0; e < 4; ++e) { o[e] = (_Float16)f0[e]; o[4 + e] = (_Float16)f1[e]; }
  *(v8h*)(out + u * 8) = o;
}

// silu(f32) -> f16, 8 elements/thread
__global__ __launch_bounds__(256)
void k_silu16(const float* __restrict__ in, _Float16* __restrict__ out, size_t n8) {
  const size_t u = (size_t)blockIdx.x * 256 + threadIdx.x;
  if (u >= n8) return;
  const float* src = in + u * 8;
  const v4f f0 = *(const v4f*)src;
  const v4f f1 = *(const v4f*)(src + 4);
  v8h o;
  #pragma unroll
  for (int e = 0; e < 4; ++e) {
    const float x0 = f0[e], x1 = f1[e];
    o[e]     = (_Float16)(x0 / (1.0f + __expf(-x0)));
    o[4 + e] = (_Float16)(x1 / (1.0f + __expf(-x1)));
  }
  *(v8h*)(out + u * 8) = o;
}

// inout += a, 4 elements/thread (n must be a multiple of 4)
__global__ __launch_bounds__(256)
void k_vadd(const float* __restrict__ a, float* __restrict__ inout, size_t n4) {
  const size_t u = (size_t)blockIdx.x * 256 + threadIdx.x;
  if (u >= n4) return;
  const v4f av = *(const v4f*)(a + u * 4);
  v4f bv = *(const v4f*)(inout + u * 4);
  #pragma unroll
  for (int e = 0; e < 4; ++e) bv[e] += av[e];
  *(v4f*)(inout + u * 4) = bv;
}

// x[S,B,E] -> xb[B,S,E] (f32) and xes[B,E,S] (f16, for the delta@x GEMM)
__global__ __launch_bounds__(256)
void k_prep(const float* __restrict__ x, float* __restrict__ xb, _Float16* __restrict__ xes) {
  const size_t idx = (size_t)blockIdx.x * 256 + threadIdx.x;
  if (idx >= (size_t)Sc * Bc * Ec) return;
  const int e = idx % Ec;
  const size_t t = idx / Ec;
  const int b = t % Bc;
  const int s = t / Bc;
  const float v = x[idx];
  xb[((size_t)b * Sc + s) * Ec + e] = v;
  xes[((size_t)b * Ec + e) * Sc + s] = (_Float16)v;
}

// delta16[b,k,l,m] = (pos[b,l,k]-pos[b,m,k]) / (dist+1), with reference masking
__global__ __launch_bounds__(256)
void k_delta(const float* __restrict__ pos, const unsigned char* __restrict__ pm,
             _Float16* __restrict__ delta16) {
  const size_t idx = (size_t)blockIdx.x * 256 + threadIdx.x;
  if (idx >= (size_t)Bc * Sc * Sc) return;
  const int m = idx % Sc;
  const int l = (idx / Sc) % Sc;
  const int b = idx / ((size_t)Sc * Sc);
  const float d0 = pos[((size_t)b * Sc + l) * 3 + 0] - pos[((size_t)b * Sc + m) * 3 + 0];
  const float d1 = pos[((size_t)b * Sc + l) * 3 + 1] - pos[((size_t)b * Sc + m) * 3 + 1];
  const float d2 = pos[((size_t)b * Sc + l) * 3 + 2] - pos[((size_t)b * Sc + m) * 3 + 2];
  const float sq = d0 * d0 + d1 * d1 + d2 * d2;
  float dist = (sq == 0.0f) ? 0.0f : sqrtf(sq);
  if (pm[b * Sc + l] || pm[b * Sc + m]) dist = 1e6f;
  const float inv = 1.0f / (dist + 1.0f);
  delta16[(((size_t)b * 3 + 0) * Sc + l) * Sc + m] = (_Float16)(d0 * inv);
  delta16[(((size_t)b * 3 + 1) * Sc + l) * Sc + m] = (_Float16)(d1 * inv);
  delta16[(((size_t)b * 3 + 2) * Sc + l) * Sc + m] = (_Float16)(d2 * inv);
}

// x = xn * norm3(ss[...,:E]) + norm3(ss[...,E:]) + x   (in-place residual update)
__global__ __launch_bounds__(256)
void k_xupdate(const float* __restrict__ xn, const float* __restrict__ ss,
               float* __restrict__ xb) {
  const size_t idx = (size_t)blockIdx.x * 256 + threadIdx.x;
  if (idx >= (size_t)Bc * Sc * Ec) return;
  const int e = idx % Ec;
  const size_t bs = idx / Ec;
  const float* r0 = ss + bs * (size_t)(6 * Ec);  // [3][2E] per (b,s)
  const float a0 = r0[e],            a1 = r0[2 * Ec + e], a2 = r0[4 * Ec + e];
  const float c0 = r0[Ec + e],       c1 = r0[3 * Ec + e], c2 = r0[5 * Ec + e];
  const float n1 = sqrtf(a0 * a0 + a1 * a1 + a2 * a2);
  const float n2 = sqrtf(c0 * c0 + c1 * c1 + c2 * c2);
  xb[idx] = xn[idx] * n1 + n2 + xb[idx];
}

// ---------------- host orchestration ----------------
extern "C" void kernel_launch(void* const* d_in, const int* in_sizes, int n_in,
                              void* d_out, int out_size, void* d_ws, size_t ws_size,
                              hipStream_t stream) {
  (void)in_sizes; (void)n_in; (void)out_size; (void)ws_size;

  const float* x_in   = (const float*)d_in[0];
  const float* pos    = (const float*)d_in[1];
  const float* gbias  = (const float*)d_in[2];
  const unsigned char* pm = (const unsigned char*)d_in[3];
  const float* ln_x_w = (const float*)d_in[4];
  const float* ln_x_b = (const float*)d_in[5];
  const float* ln_v_w = (const float*)d_in[6];
  const float* ln_v_b = (const float*)d_in[7];
  const float* ada_w     = (const float*)d_in[8];
  const float* ada_mlp_w = (const float*)d_in[9];
  const float* out_w     = (const float*)d_in[10];
  const float* xm_w = (const float*)d_in[11];
  const float* xm_b = (const float*)d_in[12];
  const float* vm_w = (const float*)d_in[13];
  const float* vm_b = (const float*)d_in[14];
  const float* mlp_w1 = (const float*)d_in[15];
  const float* mlp_w2 = (const float*)d_in[16];
  const float* fxw = (const float*)d_in[17];
  const float* fxb = (const float*)d_in[18];
  const float* fvw = (const float*)d_in[19];
  const float* fvb = (const float*)d_in[20];

  // ---- workspace bump allocator ----
  char* base = (char*)d_ws;
  size_t off = 0;
  auto alloc = [&](size_t bytes) -> void* {
    void* p = base + off;
    off = (off + bytes + 255) & ~(size_t)255;
    return p;
  };

  const size_t nX  = (size_t)Bc * Sc * Ec;          // 2M
  const size_t nV  = (size_t)Bc * Sc * 3 * Ec;      // 6.3M
  const size_t nF  = (size_t)Bc * Sc * 3 * FFNc;    // 25.2M

  _Float16* w16_out  = (_Float16*)alloc((size_t)Lc * Ec * Ec * 2);
  _Float16* w16_ada  = (_Float16*)alloc((size_t)Lc * 2 * Ec * Ec * 2);
  _Float16* w16_adam = (_Float16*)alloc((size_t)Lc * 2 * Ec * Ec * 2);
  _Float16* w16_m1   = (_Float16*)alloc((size_t)Lc * FFNc * Ec * 2);
  _Float16* w16_m2   = (_Float16*)alloc((size_t)Lc * Ec * FFNc * 2);
  float*    xb   = (float*)alloc(nX * 4);
  float*    vecb = (float*)alloc(nV * 4);
  float*    xn   = (float*)alloc(nX * 4);
  float*    vn   = (float*)alloc(nV * 4);
  float*    t32  = (float*)alloc(nV * 4);            // v1 / vm
  _Float16* A16  = (_Float16*)alloc(nF * 2);         // f16 activations (max [12288,2048])
  float*    C32  = (float*)alloc(nF * 4);            // f32 GEMM outputs (max [12288,2048])

  // delta (B*3*S*S f16) and x^T (B*E*S f16) overlay A16: consumed before layer loop
  _Float16* delta16 = A16;
  _Float16* xes16   = A16 + (size_t)Bc * 3 * Sc * Sc;

  auto blocks = [](size_t n) { return (unsigned)((n + 255) / 256); };
  auto gemm = [&](const _Float16* A, const _Float16* W, float* C,
                  int M, int N, int K, int ldc) {
    dim3 g(M / 64, N / 64);
    k_gemm<<<g, 128, 0, stream>>>(A, W, C, K, ldc);
  };
  auto cvt16 = [&](const float* src, _Float16* dst, size_t n) {
    k_copy16<<<blocks(n / 8), 256, 0, stream>>>(src, dst, n / 8);
  };
  auto silu16 = [&](const float* src, _Float16* dst, size_t n) {
    k_silu16<<<blocks(n / 8), 256, 0, stream>>>(src, dst, n / 8);
  };

  // ---- one-time prep ----
  cvt16(out_w, w16_out, (size_t)Lc * Ec * Ec);
  cvt16(ada_w, w16_ada, (size_t)Lc * 2 * Ec * Ec);
  cvt16(ada_mlp_w, w16_adam, (size_t)Lc * 2 * Ec * Ec);
  cvt16(mlp_w1, w16_m1, (size_t)Lc * FFNc * Ec);
  cvt16(mlp_w2, w16_m2, (size_t)Lc * Ec * FFNc);

  k_prep<<<blocks((size_t)Sc * Bc * Ec), 256, 0, stream>>>(x_in, xb, xes16);
  k_delta<<<blocks((size_t)Bc * Sc * Sc), 256, 0, stream>>>(pos, pm, delta16);

  // vec[b,:,k,:] = delta[b,k] (SxS) @ x[b] (SxE)  -> 12 WMMA GEMMs
  for (int b = 0; b < Bc; ++b)
    for (int kk = 0; kk < 3; ++kk)
      gemm(delta16 + ((size_t)(b * 3 + kk)) * Sc * Sc,
           xes16 + (size_t)b * Ec * Sc,
           vecb + (size_t)b * Sc * 3 * Ec + (size_t)kk * Ec,
           Sc, Ec, Sc, 3 * Ec);

  const int M3 = Bc * Sc * 3;  // 12288 rows for all [*,3,*] GEMMs

  // ---- layer loop ----
  for (int l = 0; l < Lc; ++l) {
    // attention half
    k_ln<<<Bc * Sc, 256, 0, stream>>>(xb, ln_x_w + l * Ec, ln_x_b + l * Ec, xn);
    k_ln<<<Bc * Sc * 3, 256, 0, stream>>>(vecb, ln_v_w + l * Ec, ln_v_b + l * Ec, vn);
    k_attn<<<dim3(Sc / 64, Hc, Bc), 128, 0, stream>>>(vn, gbias, pm, A16);
    gemm(A16, w16_out + (size_t)l * Ec * Ec, t32, M3, Ec, Ec, Ec);             // v1
    silu16(t32, A16, nV);
    gemm(A16, w16_ada + (size_t)l * 2 * Ec * Ec, C32, M3, 2 * Ec, Ec, 2 * Ec); // ss
    k_xupdate<<<blocks(nX), 256, 0, stream>>>(xn, C32, xb);
    k_vadd<<<blocks(nV / 4), 256, 0, stream>>>(t32, vecb, nV / 4);             // vec = v1 + rv

    // mlp half
    k_ln<<<Bc * Sc, 256, 0, stream>>>(xb, xm_w + l * Ec, xm_b + l * Ec, xn);
    k_ln<<<Bc * Sc * 3, 256, 0, stream>>>(vecb, vm_w + l * Ec, vm_b + l * Ec, vn);
    cvt16(vn, A16, nV);
    gemm(A16, w16_m1 + (size_t)l * FFNc * Ec, C32, M3, FFNc, Ec, FFNc);        // vn@W1^T
    silu16(C32, A16, nF);
    gemm(A16, w16_m2 + (size_t)l * Ec * FFNc, t32, M3, Ec, FFNc, Ec);          // vm
    silu16(t32, A16, nV);
    gemm(A16, w16_adam + (size_t)l * 2 * Ec * Ec, C32, M3, 2 * Ec, Ec, 2 * Ec);// ss2
    k_xupdate<<<blocks(nX), 256, 0, stream>>>(xn, C32, xb);
    k_vadd<<<blocks(nV / 4), 256, 0, stream>>>(t32, vecb, nV / 4);             // vec = vm + rv
  }

  // ---- final layernorms straight into d_out (x then vec, flat) ----
  float* out_f = (float*)d_out;
  k_ln<<<Bc * Sc, 256, 0, stream>>>(xb, fxw, fxb, out_f);
  k_ln<<<Bc * Sc * 3, 256, 0, stream>>>(vecb, fvw, fvb, out_f + nX);
}
